// CoAttention_6631429505130
// MI455X (gfx1250) — matrix operand order repaired and checked by reading, not compiled, not run on hardware
//
#include <hip/hip_runtime.h>
#include <stdint.h>

#define B_  64
#define H_  1024
#define LQ_ 1024
#define LA_ 1024

typedef __attribute__((ext_vector_type(16))) __bf16 v16bf;
typedef __attribute__((ext_vector_type(8)))  __bf16 v8bf;
typedef __attribute__((ext_vector_type(8)))  float  v8f;

union V16U { v16bf v; v8bf h[2]; };

// ---- order-preserving float <-> uint encoding (for atomic max on floats) ----
__device__ __forceinline__ unsigned encf(float f) {
    unsigned u = __float_as_uint(f);
    return (u & 0x80000000u) ? ~u : (u | 0x80000000u);
}
__device__ __forceinline__ float decf(unsigned u) {
    return (u & 0x80000000u) ? __uint_as_float(u & 0x7FFFFFFFu) : __uint_as_float(~u);
}

// ---- CDNA5 async global->LDS copy (16B per lane), tracked by ASYNCcnt ------
__device__ __forceinline__ void async_copy16(unsigned ldsOff, const void* g) {
    asm volatile("global_load_async_to_lds_b128 %0, %1, off"
                 :: "v"(ldsOff), "v"((unsigned long long)(uintptr_t)g)
                 : "memory");
}
__device__ __forceinline__ void wait_async0() {
    asm volatile("s_wait_asynccnt 0x0" ::: "memory");
}
// Low 32 bits of a generic LDS pointer are the LDS byte address (ISA 10.2).
__device__ __forceinline__ unsigned lds_addr(const void* p) {
    return (unsigned)(uintptr_t)p;
}

// ---------------------------------------------------------------------------
// Transpose + convert f32 [rows x cols] -> bf16 [cols x rows], per batch (z).
// ---------------------------------------------------------------------------
__global__ void k_transpose_bf16(const float* __restrict__ src, __bf16* __restrict__ dst,
                                 int rows, int cols) {
    __shared__ float tile[32][33];
    const int b = blockIdx.z;
    const float* s = src + (size_t)b * rows * cols;
    __bf16*      d = dst + (size_t)b * rows * cols;
    const int c0 = blockIdx.x * 32, r0 = blockIdx.y * 32;
    for (int i = threadIdx.y; i < 32; i += 8)
        tile[i][threadIdx.x] = s[(size_t)(r0 + i) * cols + (c0 + threadIdx.x)];
    __syncthreads();
    for (int i = threadIdx.y; i < 32; i += 8)
        d[(size_t)(c0 + i) * rows + (r0 + threadIdx.x)] = (__bf16)tile[threadIdx.x][i];
}

// ---------------------------------------------------------------------------
// bf16 WMMA GEMM: C[M,N] = Aop[M,K] * Bop[N,K]^T   (both row-major, ld = K)
// MODE 0: store C as bf16 (ld = N).
// MODE 1: C -> tanh(C); atomic-max row maxima (over N) and col maxima (over M).
// Workgroup tile 128(M) x 128(N), BK=32, 256 threads = 8 waves (2M x 4N),
// each wave computes 64x32 via 4x2 accumulators. Double-buffered LDS filled
// with global_load_async_to_lds_b128, one barrier per K-step.
// ---------------------------------------------------------------------------
#define SPITCH 40   // LDS row pitch in bf16 elems (80B) -> conflict-free b128 reads
#define BK     32

template <int MODE>
__global__ void k_gemm(const __bf16* __restrict__ Aop, size_t aStride,
                       const __bf16* __restrict__ Bop, size_t bStride,
                       __bf16* __restrict__ Cout, size_t cStride,
                       unsigned* __restrict__ rowMax, unsigned* __restrict__ colMax,
                       int M, int N, int K) {
    const int b = blockIdx.z;
    const __bf16* Ag = Aop + (size_t)b * aStride;
    const __bf16* Bg = Bop + (size_t)b * bStride;
    const int m0 = blockIdx.y * 128;
    const int n0 = blockIdx.x * 128;

    __shared__ __align__(16) __bf16 sA[2][128 * SPITCH];
    __shared__ __align__(16) __bf16 sB[2][128 * SPITCH];

    const int tid  = threadIdx.x;
    const int wave = tid >> 5, lane = tid & 31;
    const int wm = wave & 1;       // 2 waves along M, 64 rows each
    const int wn = wave >> 1;      // 4 waves along N, 32 cols each
    const int r16 = lane & 15;     // A row / B col within 16
    const int kg  = lane >> 4;     // lane-group (K half select)

    v8f acc[4][2];
#pragma unroll
    for (int ms = 0; ms < 4; ++ms)
#pragma unroll
        for (int ns = 0; ns < 2; ++ns)
#pragma unroll
            for (int j = 0; j < 8; ++j) acc[ms][ns][j] = 0.0f;

    // stage one 128x32 A panel + 128x32 B panel into buffer `buf` (async DMA)
    auto stage = [&](int buf, int k0) {
#pragma unroll
        for (int t = tid; t < 512; t += 256) {
            const int row = t >> 2, seg = (t & 3) * 8;
            async_copy16(lds_addr(&sA[buf][row * SPITCH + seg]),
                         &Ag[(size_t)(m0 + row) * K + k0 + seg]);
            async_copy16(lds_addr(&sB[buf][row * SPITCH + seg]),
                         &Bg[(size_t)(n0 + row) * K + k0 + seg]);
        }
    };

    stage(0, 0);
    wait_async0();
    __syncthreads();

    int pBuf = 0;
    for (int k0 = 0; k0 < K; k0 += BK) {
        const bool more = (k0 + BK) < K;
        if (more) stage(pBuf ^ 1, k0 + BK);   // async fill of next buffer

        const __bf16* bufA = sA[pBuf];
        const __bf16* bufB = sB[pBuf];

        V16U af[4];
#pragma unroll
        for (int ms = 0; ms < 4; ++ms) {
            const __bf16* ap = &bufA[(wm * 64 + ms * 16 + r16) * SPITCH];
            af[ms].h[0] = *(const v8bf*)(ap + kg * 8);
            af[ms].h[1] = *(const v8bf*)(ap + 16 + kg * 8);
        }
        V16U bfr[2];
#pragma unroll
        for (int ns = 0; ns < 2; ++ns) {
            const __bf16* bp = &bufB[(wn * 32 + ns * 16 + r16) * SPITCH];
            bfr[ns].h[0] = *(const v8bf*)(bp + kg * 8);
            bfr[ns].h[1] = *(const v8bf*)(bp + 16 + kg * 8);
        }
#pragma unroll
        for (int ms = 0; ms < 4; ++ms)
#pragma unroll
            for (int ns = 0; ns < 2; ++ns)
                acc[ms][ns] = __builtin_amdgcn_wmma_f32_16x16x32_bf16(
                    false, af[ms].v, false, bfr[ns].v, (short)0, acc[ms][ns], false, false);

        if (more) wait_async0();   // DMA into next buffer complete (this wave)
        __syncthreads();           // all waves: reads of pBuf done, fills visible
        pBuf ^= 1;
    }

    if (MODE == 0) {
        __bf16* C = Cout + (size_t)b * cStride;
#pragma unroll
        for (int ms = 0; ms < 4; ++ms)
#pragma unroll
            for (int ns = 0; ns < 2; ++ns) {
                const int n  = n0 + wn * 32 + ns * 16 + r16;
                const int mb = m0 + wm * 64 + ms * 16 + kg * 8;
#pragma unroll
                for (int j = 0; j < 8; ++j)
                    C[(size_t)(mb + j) * N + n] = (__bf16)acc[ms][ns][j];
            }
    } else {
        unsigned* rm = rowMax + (size_t)b * M;
        unsigned* cm = colMax + (size_t)b * N;
#pragma unroll
        for (int ms = 0; ms < 4; ++ms)
#pragma unroll
            for (int ns = 0; ns < 2; ++ns)
#pragma unroll
                for (int j = 0; j < 8; ++j)
                    acc[ms][ns][j] = tanhf(acc[ms][ns][j]);
        // column maxima: lane holds col (wn,ns,r16); fold rows then lane halves
#pragma unroll
        for (int ns = 0; ns < 2; ++ns) {
            float cmx = -3.4e38f;
#pragma unroll
            for (int ms = 0; ms < 4; ++ms)
#pragma unroll
                for (int j = 0; j < 8; ++j) cmx = fmaxf(cmx, acc[ms][ns][j]);
            cmx = fmaxf(cmx, __shfl_xor(cmx, 16));
            if (kg == 0) atomicMax(&cm[n0 + wn * 32 + ns * 16 + r16], encf(cmx));
        }
        // row maxima: reg j = row (kg*8+j) of subtile ms; reduce across 16 N-lanes
#pragma unroll
        for (int ms = 0; ms < 4; ++ms)
#pragma unroll
            for (int j = 0; j < 8; ++j) {
                float r = fmaxf(acc[ms][0][j], acc[ms][1][j]);
                r = fmaxf(r, __shfl_xor(r, 8));
                r = fmaxf(r, __shfl_xor(r, 4));
                r = fmaxf(r, __shfl_xor(r, 2));
                r = fmaxf(r, __shfl_xor(r, 1));
                if (r16 == 0) atomicMax(&rm[m0 + wm * 64 + ms * 16 + kg * 8 + j], encf(r));
            }
    }
}

// ---------------------------------------------------------------------------
__global__ void k_fill_u32(unsigned* __restrict__ p, unsigned v, int n) {
    int i = blockIdx.x * 256 + threadIdx.x;
    if (i < n) p[i] = v;
}

// per-batch softmax over L encoded maxima -> float pool weights
__global__ void k_softmax(const unsigned* __restrict__ encIn, float* __restrict__ pool, int L) {
    __shared__ float red[256];
    const int b = blockIdx.x, tid = threadIdx.x;
    const unsigned* src = encIn + (size_t)b * L;
    float* dst = pool + (size_t)b * L;

    float lmax = -3.4e38f;
    for (int i = tid; i < L; i += 256) lmax = fmaxf(lmax, decf(src[i]));
    red[tid] = lmax; __syncthreads();
    for (int s = 128; s > 0; s >>= 1) {
        if (tid < s) red[tid] = fmaxf(red[tid], red[tid + s]);
        __syncthreads();
    }
    const float m = red[0];
    __syncthreads();

    float lsum = 0.0f;
    for (int i = tid; i < L; i += 256) {
        float e = __expf(decf(src[i]) - m);
        dst[i] = e;
        lsum += e;
    }
    red[tid] = lsum; __syncthreads();
    for (int s = 128; s > 0; s >>= 1) {
        if (tid < s) red[tid] += red[tid + s];
        __syncthreads();
    }
    const float inv = 1.0f / red[0];
    __syncthreads();
    for (int i = tid; i < L; i += 256) dst[i] *= inv;
}

// out[b,h] = sum_i X[b,h,i] * pool[b,i]
__global__ void k_wsum(const float* __restrict__ X, const float* __restrict__ pool,
                       float* __restrict__ out, int L) {
    __shared__ float red[256];
    const int h = blockIdx.x, b = blockIdx.y, tid = threadIdx.x;
    const float* x = X + ((size_t)b * H_ + h) * L;
    const float* p = pool + (size_t)b * L;
    float s = 0.0f;
    for (int i = tid; i < L; i += 256) s += x[i] * p[i];
    red[tid] = s; __syncthreads();
    for (int t = 128; t > 0; t >>= 1) {
        if (tid < t) red[tid] += red[tid + t];
        __syncthreads();
    }
    if (tid == 0) out[(size_t)b * H_ + h] = red[0];
}

// ---------------------------------------------------------------------------
extern "C" void kernel_launch(void* const* d_in, const int* in_sizes, int n_in,
                              void* d_out, int out_size, void* d_ws, size_t ws_size,
                              hipStream_t stream) {
    const float* Q = (const float*)d_in[0];   // [B,H,LQ]
    const float* A = (const float*)d_in[1];   // [B,H,LA]
    const float* U = (const float*)d_in[2];   // [H,H]
    float* out = (float*)d_out;               // rq[B,H] ++ ra[B,H]

    char* w = (char*)d_ws;
    size_t off = 0;
    auto take = [&](size_t bytes) -> void* {
        void* p = w + off;
        off += (bytes + 255) & ~(size_t)255;
        return p;
    };
    __bf16*   Qt     = (__bf16*)  take((size_t)B_ * LQ_ * H_ * 2); // Q^T per batch [LQ,H]
    __bf16*   At     = (__bf16*)  take((size_t)B_ * LA_ * H_ * 2); // A^T per batch [LA,H]
    __bf16*   Ut     = (__bf16*)  take((size_t)H_ * H_ * 2);       // U^T [H,H]
    __bf16*   QU     = (__bf16*)  take((size_t)B_ * LQ_ * H_ * 2); // [LQ,H] per batch
    unsigned* rowmax = (unsigned*)take((size_t)B_ * LQ_ * 4);
    unsigned* colmax = (unsigned*)take((size_t)B_ * LA_ * 4);
    float*    qpool  = (float*)   take((size_t)B_ * LQ_ * 4);
    float*    apool  = (float*)   take((size_t)B_ * LA_ * 4);

    const dim3 tb(32, 8, 1);
    k_transpose_bf16<<<dim3(LQ_ / 32, H_ / 32, B_), tb, 0, stream>>>(Q, Qt, H_, LQ_);
    k_transpose_bf16<<<dim3(LA_ / 32, H_ / 32, B_), tb, 0, stream>>>(A, At, H_, LA_);
    k_transpose_bf16<<<dim3(H_ / 32, H_ / 32, 1), tb, 0, stream>>>(U, Ut, H_, H_);

    // GEMM1: QU[b] = Qt[b] (M=LQ,K=H) x Ut^T (B-op [N=H,K=H])
    k_gemm<0><<<dim3(H_ / 128, LQ_ / 128, B_), 256, 0, stream>>>(
        Qt, (size_t)LQ_ * H_, Ut, (size_t)0, QU, (size_t)LQ_ * H_,
        nullptr, nullptr, LQ_, H_, H_);

    k_fill_u32<<<(B_ * LQ_ + 255) / 256, 256, 0, stream>>>(rowmax, 0u, B_ * LQ_);
    k_fill_u32<<<(B_ * LA_ + 255) / 256, 256, 0, stream>>>(colmax, 0u, B_ * LA_);

    // GEMM2: G = tanh(QU[b] x At[b]^T), fused max-reductions
    k_gemm<1><<<dim3(LA_ / 128, LQ_ / 128, B_), 256, 0, stream>>>(
        QU, (size_t)LQ_ * H_, At, (size_t)LA_ * H_, nullptr, (size_t)0,
        rowmax, colmax, LQ_, LA_, H_);

    k_softmax<<<B_, 256, 0, stream>>>(rowmax, qpool, LQ_);
    k_softmax<<<B_, 256, 0, stream>>>(colmax, apool, LA_);

    k_wsum<<<dim3(H_, B_), 256, 0, stream>>>(Q, qpool, out, LQ_);
    k_wsum<<<dim3(H_, B_), 256, 0, stream>>>(A, apool, out + (size_t)B_ * H_, LA_);
}
